// FPChainTreeLSTM_35940286333321
// MI455X (gfx1250) — matrix operand order repaired and checked by reading, not compiled, not run on hardware
//
#include <hip/hip_runtime.h>
#include <stdint.h>

// ---------------------------------------------------------------------------
// Types for WMMA (CDNA5 / gfx1250, wave32)
// ---------------------------------------------------------------------------
typedef __attribute__((ext_vector_type(16))) __bf16 v16bf;
typedef __attribute__((ext_vector_type(8)))  float  v8f;

struct alignas(16) Q16  { unsigned x, y, z, w; };
struct alignas(32) Q16x2 { Q16 lo, hi; };
struct alignas(16) F4   { float x, y, z, w; };

// ---------------------------------------------------------------------------
// Problem constants
// ---------------------------------------------------------------------------
#define IN_DIM 1024
#define MEM    1024
#define NNODE  2048
#define NGX    (5 * MEM)   // 5120
#define NGH    (4 * MEM)   // 4096
#define NBLK   16          // recurrence workgroups == max cluster size
#define TPB    1024

// Workspace layout (bytes)
constexpr size_t OFF_INBF = 0;                                          // 2048x1024 bf16
constexpr size_t OFF_WXT  = OFF_INBF + (size_t)NNODE * IN_DIM * 2;      // 5120x1024 bf16
constexpr size_t OFF_WHT  = OFF_WXT  + (size_t)NGX   * IN_DIM * 2;      // 4096x1024 bf16
constexpr size_t OFF_WUMT = OFF_WHT  + (size_t)NGH   * MEM    * 2;      // 1024x1024 bf16
constexpr size_t OFF_GX   = OFF_WUMT + (size_t)MEM   * MEM    * 2;      // 2048x5120 f32
constexpr size_t OFF_H    = OFF_GX   + (size_t)NNODE * NGX    * 4;      // 1024 f32
constexpr size_t OFF_ZT   = OFF_H    + (size_t)MEM * 4;                 // 1024 f32
constexpr size_t OFF_BAR  = OFF_ZT   + (size_t)MEM * 4;                 // 2 u32

// ---------------------------------------------------------------------------
// Helpers
// ---------------------------------------------------------------------------
__device__ __forceinline__ unsigned short f2bf(float f) {
  unsigned u = __float_as_uint(f);
  u += 0x7fffu + ((u >> 16) & 1u);          // round-to-nearest-even
  return (unsigned short)(u >> 16);
}
__device__ __forceinline__ float bflo(unsigned u) { return __uint_as_float(u << 16); }
__device__ __forceinline__ float bfhi(unsigned u) { return __uint_as_float(u & 0xffff0000u); }
__device__ __forceinline__ float sigf(float x)    { return 1.0f / (1.0f + __expf(-x)); }

// dot of n8*8 bf16 weights (global/L2) with f32 vector (LDS)
__device__ __forceinline__ float dot_bf16_f32(const unsigned short* __restrict__ w,
                                              const float* __restrict__ x, int n8) {
  const Q16* wq = (const Q16*)w;
  float s = 0.0f;
#pragma unroll 8
  for (int t = 0; t < n8; ++t) {
    Q16 q = wq[t];
    const F4* xp = (const F4*)(x + t * 8);
    F4 x0 = xp[0], x1 = xp[1];
    s = fmaf(bflo(q.x), x0.x, s); s = fmaf(bfhi(q.x), x0.y, s);
    s = fmaf(bflo(q.y), x0.z, s); s = fmaf(bfhi(q.y), x0.w, s);
    s = fmaf(bflo(q.z), x1.x, s); s = fmaf(bfhi(q.z), x1.y, s);
    s = fmaf(bflo(q.w), x1.z, s); s = fmaf(bfhi(q.w), x1.w, s);
  }
  return s;
}

// ---------------------------------------------------------------------------
// Prep kernels: cast inputs to bf16; transpose+cast weights to k-contiguous bf16
// ---------------------------------------------------------------------------
__global__ void k_cast_bf16(const float* __restrict__ in, unsigned short* __restrict__ out, int n) {
  int i = blockIdx.x * blockDim.x + threadIdx.x;
  if (i < n) out[i] = f2bf(in[i]);
}

// in: R x C f32 (row-major) -> out: C x R bf16 (row-major)   R,C multiples of 32
__global__ void k_transpose_bf16(const float* __restrict__ in, unsigned short* __restrict__ out,
                                 int R, int C) {
  __shared__ float tile[32][33];
  int c0 = blockIdx.x * 32, r0 = blockIdx.y * 32;
  for (int i = threadIdx.y; i < 32; i += 8)
    tile[i][threadIdx.x] = in[(size_t)(r0 + i) * C + c0 + threadIdx.x];
  __syncthreads();
  for (int i = threadIdx.y; i < 32; i += 8)
    out[(size_t)(c0 + i) * R + r0 + threadIdx.x] = f2bf(tile[threadIdx.x][i]);
}

__global__ void k_init(float* __restrict__ h_g, float* __restrict__ zt_g,
                       unsigned* __restrict__ bar) {
  int t = blockIdx.x * blockDim.x + threadIdx.x;
  if (t < MEM) { h_g[t] = 0.0f; zt_g[t] = 0.0f; }
  if (t == 0)  { bar[0] = 0u; bar[1] = 0u; }
}

// ---------------------------------------------------------------------------
// Phase 1: Gx = inputs @ Wx + bx   via v_wmma_f32_16x16x32_bf16
//   block = 256 threads (8 waves), block tile 128(M) x 64(N), wave tile 16x64
//   A: 2048x1024 bf16 (k-contiguous), BT: 5120x1024 bf16 (k-contiguous)
// ---------------------------------------------------------------------------
__global__ __launch_bounds__(256)
void k_gemm_gx(const unsigned short* __restrict__ A,
               const unsigned short* __restrict__ BT,
               const float* __restrict__ bias,
               float* __restrict__ Gx) {
  const int K = IN_DIM;
  int lane = threadIdx.x & 31;
  int wave = threadIdx.x >> 5;                 // 0..7
  int half = lane >> 4, r = lane & 15;
  int m0 = blockIdx.y * 128 + wave * 16;
  int n0 = blockIdx.x * 64;

  v8f acc[4] = {};
  const Q16* arow = (const Q16*)(A + (size_t)(m0 + r) * K);
  const Q16* brow[4];
#pragma unroll
  for (int nt = 0; nt < 4; ++nt)
    brow[nt] = (const Q16*)(BT + (size_t)(n0 + nt * 16 + r) * K);

#pragma unroll 4
  for (int k0 = 0; k0 < K; k0 += 32) {
    // A fragment (16-bit A 16x32 layout): lanes 0-15 hold K[0..7],K[16..23]
    Q16x2 araw;
    araw.lo = arow[(k0 >> 3) + half];
    araw.hi = arow[(k0 >> 3) + 2 + half];
    v16bf av = __builtin_bit_cast(v16bf, araw);
#pragma unroll
    for (int nt = 0; nt < 4; ++nt) {
      // B fragment (32x16): lanes 0-15 = N, K[0..7],K[16..23]; lanes 16-31 = K[8..15],K[24..31]
      Q16x2 braw;
      braw.lo = brow[nt][(k0 >> 3) + half];
      braw.hi = brow[nt][(k0 >> 3) + 2 + half];
      v16bf bv = __builtin_bit_cast(v16bf, braw);
      acc[nt] = __builtin_amdgcn_wmma_f32_16x16x32_bf16(
          false, av, false, bv, (short)0, acc[nt], false, false);
    }
  }

#pragma unroll
  for (int nt = 0; nt < 4; ++nt) {
    int col = n0 + nt * 16 + r;
    float bv = bias[col];
#pragma unroll
    for (int v = 0; v < 8; ++v) {
      int row = m0 + half * 8 + v;            // C/D layout: VGPR v -> M = half*8+v
      Gx[(size_t)row * NGX + col] = acc[nt][v] + bv;
    }
  }
}

// ---------------------------------------------------------------------------
// Grid-wide sync for the recurrence.
// If the 16 workgroups were dispatched as one cluster (CDNA5), use the HW
// cluster split-barrier: one wave signals (ISA: member count = #workgroups),
// all waves wait. Otherwise fall back to an L2-atomic sense barrier.
// ---------------------------------------------------------------------------
__device__ __forceinline__ void grid_sync(unsigned* cnt, unsigned* gen) {
  __syncthreads();
  if (__builtin_amdgcn_cluster_id_x() != 0) {
    __threadfence();
    if (threadIdx.x < 32u)
      asm volatile("s_barrier_signal -3" ::: "memory");
    asm volatile("s_barrier_wait -3" ::: "memory");
  } else if (threadIdx.x == 0u) {
    __threadfence();
    unsigned g = __hip_atomic_load(gen, __ATOMIC_RELAXED, __HIP_MEMORY_SCOPE_AGENT);
    if (__hip_atomic_fetch_add(cnt, 1u, __ATOMIC_ACQ_REL, __HIP_MEMORY_SCOPE_AGENT) == NBLK - 1u) {
      __hip_atomic_store(cnt, 0u, __ATOMIC_RELAXED, __HIP_MEMORY_SCOPE_AGENT);
      __hip_atomic_store(gen, g + 1u, __ATOMIC_RELEASE, __HIP_MEMORY_SCOPE_AGENT);
    } else {
      while (__hip_atomic_load(gen, __ATOMIC_ACQUIRE, __HIP_MEMORY_SCOPE_AGENT) == g)
        __builtin_amdgcn_s_sleep(2);
    }
  }
  __syncthreads();
}

// ---------------------------------------------------------------------------
// Phase 2: 2048-step sequential chain. 16 blocks x 1024 threads.
// Block b owns elements j in [b*64, b*64+64): its c stays in LDS forever,
// its slice of h/zt goes through L2. Weights (bf16, 10MB) are L2-resident.
// ---------------------------------------------------------------------------
__global__ __launch_bounds__(TPB, 1)
void k_recurrence(const float* __restrict__ Gx,               // 2048 x 5120
                  const unsigned short* __restrict__ WhT,     // 4096 x 1024 bf16
                  const float* __restrict__ bh,               // 4096
                  const unsigned short* __restrict__ WumT,    // 1024 x 1024 bf16
                  const float* __restrict__ bum,              // 1024
                  const float* __restrict__ pic, const float* __restrict__ pfc,
                  const float* __restrict__ poc, const float* __restrict__ pzc,
                  float* __restrict__ h_g, float* __restrict__ zt_g,
                  unsigned* __restrict__ bar_cnt, unsigned* __restrict__ bar_gen,
                  float* __restrict__ out) {
  __shared__ __align__(16) float h_s[MEM];
  __shared__ __align__(16) float zt_s[MEM];
  __shared__ float gh_s[256];
  __shared__ float c_own[64], i_s[64], f_s[64], oh_s[64], m_s[64], hmax_s[64];

  const int tid = threadIdx.x;
  const int blk = blockIdx.x;

  if (tid < 64) { c_own[tid] = 0.0f; hmax_s[tid] = -1e30f; }
  __syncthreads();

  // gh phase mapping: 256 outputs x 4-way split-K (1024 threads exactly)
  const int oA = tid >> 2, pA = tid & 3;         // output 0..255, part 0..3
  const int gate = oA >> 6, jjA = oA & 63;
  const int rowA = gate * MEM + blk * 64 + jjA;
  const unsigned short* whrow = WhT + (size_t)rowA * MEM + pA * 256;

  // Wum phase mapping: 64 outputs x 16-way split-K
  const int oB = tid >> 4, pB = tid & 15;        // output 0..63, part 0..15
  const int jB = blk * 64 + oB;
  const unsigned short* wumrow = WumT + (size_t)jB * MEM + pB * 64;

  for (int step = 0; step < NNODE; ++step) {
    const float* gx = Gx + (size_t)step * NGX;

    // ---- stage parent h into LDS
    h_s[tid] = h_g[tid];
    __syncthreads();

    // ---- gh = h_p @ Wh + bh  (this block: its 4x64 outputs)
    {
      float s = dot_bf16_f32(whrow, h_s + pA * 256, 32);
      s += __shfl_xor(s, 1, 32);
      s += __shfl_xor(s, 2, 32);
      if (pA == 0) gh_s[oA] = s + bh[rowA];
    }
    __syncthreads();

    // ---- gates i,f,z (owned 64 elements); publish zt = z*tanh(c_p)
    if (tid < 64) {
      int jj = tid, j = blk * 64 + jj;
      float cp = c_own[jj];
      float iv = sigf(gx[j]            + gh_s[jj]        + pic[j] * cp);
      float fv = sigf(gx[2 * MEM + j]  + gh_s[128 + jj]  + pfc[j] * cp);
      float zv = sigf(gx[3 * MEM + j]  + gh_s[192 + jj]  + pzc[j] * cp);
      i_s[jj] = iv; f_s[jj] = fv; oh_s[jj] = gh_s[64 + jj];
      zt_g[j] = zv * tanhf(cp);
    }
    grid_sync(bar_cnt, bar_gen);               // zt globally visible

    // ---- m = zt @ Wum + bum  (owned 64 outputs)
    zt_s[tid] = zt_g[tid];
    __syncthreads();
    {
      float s = dot_bf16_f32(wumrow, zt_s + pB * 64, 8);
      s += __shfl_xor(s, 1, 32);
      s += __shfl_xor(s, 2, 32);
      s += __shfl_xor(s, 4, 32);
      s += __shfl_xor(s, 8, 32);
      if (pB == 0) m_s[oB] = s + bum[jB];
    }
    __syncthreads();

    // ---- u, c, o, h, running max (owned 64 elements)
    if (tid < 64) {
      int jj = tid, j = blk * 64 + jj;
      float u  = tanhf(gx[4 * MEM + j] + m_s[jj]);
      float cp = c_own[jj];
      float c  = i_s[jj] * u + f_s[jj] * cp;
      float ov = sigf(gx[MEM + j] + oh_s[jj] + poc[j] * c);
      float h  = ov * tanhf(c);
      c_own[jj] = c;
      h_g[j] = h;
      hmax_s[jj] = fmaxf(hmax_s[jj], h);
    }
    grid_sync(bar_cnt, bar_gen);               // h globally visible
  }

  if (tid < 64) out[blk * 64 + tid] = hmax_s[tid];
}

// ---------------------------------------------------------------------------
// Launch
// ---------------------------------------------------------------------------
extern "C" void kernel_launch(void* const* d_in, const int* in_sizes, int n_in,
                              void* d_out, int out_size, void* d_ws, size_t ws_size,
                              hipStream_t stream) {
  (void)in_sizes; (void)n_in; (void)out_size; (void)ws_size;
  const float* inputs = (const float*)d_in[0];
  const float* Wx     = (const float*)d_in[1];
  const float* bx     = (const float*)d_in[2];
  const float* Wh     = (const float*)d_in[3];
  const float* bh     = (const float*)d_in[4];
  const float* Wum    = (const float*)d_in[5];
  const float* bum    = (const float*)d_in[6];
  const float* pic    = (const float*)d_in[7];
  const float* pfc    = (const float*)d_in[8];
  const float* poc    = (const float*)d_in[9];
  const float* pzc    = (const float*)d_in[10];

  char* ws = (char*)d_ws;
  unsigned short* inbf = (unsigned short*)(ws + OFF_INBF);
  unsigned short* WxT  = (unsigned short*)(ws + OFF_WXT);
  unsigned short* WhT  = (unsigned short*)(ws + OFF_WHT);
  unsigned short* WumT = (unsigned short*)(ws + OFF_WUMT);
  float* Gx   = (float*)(ws + OFF_GX);
  float* h_g  = (float*)(ws + OFF_H);
  float* zt_g = (float*)(ws + OFF_ZT);
  unsigned* bar = (unsigned*)(ws + OFF_BAR);

  // prep: bf16 cast + k-contiguous transposes (all weights then live in L2)
  k_cast_bf16<<<(NNODE * IN_DIM) / 256, 256, 0, stream>>>(inputs, inbf, NNODE * IN_DIM);
  dim3 tb(32, 8);
  k_transpose_bf16<<<dim3(NGX / 32, IN_DIM / 32), tb, 0, stream>>>(Wx,  WxT,  IN_DIM, NGX);
  k_transpose_bf16<<<dim3(NGH / 32, MEM    / 32), tb, 0, stream>>>(Wh,  WhT,  MEM,    NGH);
  k_transpose_bf16<<<dim3(MEM / 32, MEM    / 32), tb, 0, stream>>>(Wum, WumT, MEM,    MEM);
  k_init<<<4, 256, 0, stream>>>(h_g, zt_g, bar);

  // phase 1: batched input projection via WMMA
  k_gemm_gx<<<dim3(NGX / 64, NNODE / 128), 256, 0, stream>>>(inbf, WxT, bx, Gx);

  // phase 2: sequential chain, 16 WGs (cluster-sized) with HW/atomic grid sync
  k_recurrence<<<NBLK, TPB, 0, stream>>>(Gx, WhT, bh, WumT, bum,
                                         pic, pfc, poc, pzc,
                                         h_g, zt_g, bar, bar + 1, (float*)d_out);
}
